// QuantizedLinear_20383914787205
// MI455X (gfx1250) — matrix operand order repaired and checked by reading, not compiled, not run on hardware
//
#include <hip/hip_runtime.h>
#include <stdint.h>

typedef __attribute__((ext_vector_type(16))) _Float16 v16h;
typedef __attribute__((ext_vector_type(8)))  _Float16 v8h;
typedef __attribute__((ext_vector_type(4)))  _Float16 v4h;
typedef __attribute__((ext_vector_type(8)))  float    v8f;
typedef int v4i_gcc __attribute__((vector_size(16)));   // matches builtin param type

union FragH { v16h v; v8h h[2]; };

#define TM 128      // block tile M
#define TN 128      // block tile N
#define TK 64       // K step per LDS stage
#define LDST 72     // LDS row stride in f16 (64 + 8 pad, keeps 16B alignment)

#define AS1 __attribute__((address_space(1)))
#define AS3 __attribute__((address_space(3)))

#if defined(__HIP_DEVICE_COMPILE__) && __has_builtin(__builtin_amdgcn_global_load_async_to_lds_b128)
#define HAS_ASYNC_LDS 1
#else
#define HAS_ASYNC_LDS 0
#endif

// ---------------- shared WMMA micro-kernel (2 M-frags x 4 N-frags per wave) --------------
__device__ __forceinline__ void wave_mma_step(const _Float16 (*As)[LDST],
                                              const _Float16 (*Bs)[LDST],
                                              v8f (&acc)[2][4], int waveM, int waveN,
                                              int half, int l15) {
#pragma unroll
  for (int kk = 0; kk < TK; kk += 32) {
    FragH a[2], b[4];
    // A 16x32: lanes<16 hold K {0..7,16..23}, lanes>=16 hold K {8..15,24..31}
    const int aCol = kk + (half << 3);
#pragma unroll
    for (int mt = 0; mt < 2; ++mt) {
      const _Float16* p = &As[waveM + mt * 16 + l15][aCol];
      a[mt].h[0] = *(const v8h*)p;
      a[mt].h[1] = *(const v8h*)(p + 16);
    }
    // B 32x16: lane = column n; lanes<16 hold K {0..7,16..23}, lanes>=16 hold K {8..15,24..31}
    const int bCol = kk + (half << 3);
#pragma unroll
    for (int nt = 0; nt < 4; ++nt) {
      const _Float16* p = &Bs[waveN + nt * 16 + l15][bCol];
      b[nt].h[0] = *(const v8h*)p;
      b[nt].h[1] = *(const v8h*)(p + 16);
    }
#pragma unroll
    for (int mt = 0; mt < 2; ++mt)
#pragma unroll
      for (int nt = 0; nt < 4; ++nt)
        acc[mt][nt] = __builtin_amdgcn_wmma_f32_16x16x32_f16(
            false, a[mt].v, false, b[nt].v, (short)0, acc[mt][nt], false, false);
  }
}

__device__ __forceinline__ void store_epilogue(float* __restrict__ out,
                                               const float* __restrict__ scale,
                                               const float* __restrict__ bias,
                                               v8f (&acc)[2][4], int mBlock, int nBlock,
                                               int waveM, int waveN, int half, int l15, int N) {
  const float s = scale[0];
#pragma unroll
  for (int nt = 0; nt < 4; ++nt) {
    const int n = nBlock + waveN + nt * 16 + l15;
    const float bv = bias[n];
#pragma unroll
    for (int mt = 0; mt < 2; ++mt) {
      const int mBase = mBlock + waveM + mt * 16 + half * 8;  // C: VGPR r -> M=r (+8 hi lanes)
#pragma unroll
      for (int r = 0; r < 8; ++r)
        out[(size_t)(mBase + r) * N + n] = acc[mt][nt][r] * s + bv;
    }
  }
}

// ---------------- one-time conversion passes into workspace -----------------------------
__global__ __launch_bounds__(256) void convert_w_f16(const int* __restrict__ w,
                                                     _Float16* __restrict__ wh, long n4) {
  const long i = (long)blockIdx.x * blockDim.x + threadIdx.x;
  if (i >= n4) return;
  const int4 v = ((const int4*)w)[i];
  v4h h = { (_Float16)(float)v.x, (_Float16)(float)v.y,
            (_Float16)(float)v.z, (_Float16)(float)v.w };
  ((v4h*)wh)[i] = h;
}

__global__ __launch_bounds__(256) void convert_x_f16(const float* __restrict__ x,
                                                     _Float16* __restrict__ xh, long n4) {
  const long i = (long)blockIdx.x * blockDim.x + threadIdx.x;
  if (i >= n4) return;
  const float4 v = ((const float4*)x)[i];
  v4h h = { (_Float16)v.x, (_Float16)v.y, (_Float16)v.z, (_Float16)v.w };
  ((v4h*)xh)[i] = h;
}

// ---------------- main GEMM on pre-converted f16 operands -------------------------------
// Staging is a pure 128-bit copy global->LDS; async path keeps the inter-barrier
// region empty (only s_wait_asynccnt + one barrier per K step).
__global__ __launch_bounds__(256)
void qlinear_wmma_f16_pre(const _Float16* __restrict__ xh, const _Float16* __restrict__ wh,
                          const float* __restrict__ scale, const float* __restrict__ bias,
                          float* __restrict__ out, int M, int N, int K) {
  __shared__ _Float16 ldsA[2][TM][LDST];
  __shared__ _Float16 ldsB[2][TN][LDST];

  const int tid   = threadIdx.x;
  const int lane  = tid & 31;
  const int wave  = tid >> 5;
  const int waveM = (wave & 3) * 32;
  const int waveN = (wave >> 2) * 64;
  const int half  = lane >> 4;
  const int l15   = lane & 15;
  const int mBlock = blockIdx.y * TM;
  const int nBlock = blockIdx.x * TN;

  v8f acc[2][4] = {};
  const int NT = K / TK;
  const int colS = (tid & 7) << 3;   // f16 column within tile row (16B granules)
  const int rowS = tid >> 3;         // base row, +32 per chunk

#if HAS_ASYNC_LDS
  auto stageAsync = [&](int kt, int buf) {
    const int k0 = kt * TK;
#pragma unroll
    for (int i = 0; i < 4; ++i) {
      const int row = rowS + i * 32;
      __builtin_amdgcn_global_load_async_to_lds_b128(
          (AS1 v4i_gcc*)(uintptr_t)(xh + (size_t)(mBlock + row) * K + k0 + colS),
          (AS3 v4i_gcc*)(uintptr_t)&ldsA[buf][row][colS], 0, 0);
      __builtin_amdgcn_global_load_async_to_lds_b128(
          (AS1 v4i_gcc*)(uintptr_t)(wh + (size_t)(nBlock + row) * K + k0 + colS),
          (AS3 v4i_gcc*)(uintptr_t)&ldsB[buf][row][colS], 0, 0);
    }
  };
  auto asyncWait = [&]() {
#if __has_builtin(__builtin_amdgcn_s_wait_asynccnt)
    __builtin_amdgcn_s_wait_asynccnt(0);
#else
    asm volatile("s_wait_asynccnt 0x0" ::: "memory");
#endif
  };

  stageAsync(0, 0);
  asyncWait();
  __syncthreads();
  for (int kt = 0; kt < NT; ++kt) {
    if (kt + 1 < NT) stageAsync(kt + 1, (kt + 1) & 1);   // DMA next tile during WMMA
    if (kt + 2 < NT) {                                    // L2 prefetch two tiles ahead
      const int k0 = (kt + 2) * TK;
      if (tid < 128) __builtin_prefetch(xh + (size_t)(mBlock + tid)       * K + k0, 0, 3);
      else           __builtin_prefetch(wh + (size_t)(nBlock + tid - 128) * K + k0, 0, 3);
    }
    wave_mma_step(ldsA[kt & 1], ldsB[kt & 1], acc, waveM, waveN, half, l15);
    asyncWait();
    __syncthreads();
  }
#else
  v8h xr[4], wr[4];
  auto loadTiles = [&](int kt) {
    const int k0 = kt * TK;
#pragma unroll
    for (int i = 0; i < 4; ++i) {
      const int row = rowS + i * 32;
      xr[i] = *(const v8h*)(xh + (size_t)(mBlock + row) * K + k0 + colS);
      wr[i] = *(const v8h*)(wh + (size_t)(nBlock + row) * K + k0 + colS);
    }
  };
  auto stageTiles = [&](int buf) {
#pragma unroll
    for (int i = 0; i < 4; ++i) {
      const int row = rowS + i * 32;
      *(v8h*)&ldsA[buf][row][colS] = xr[i];
      *(v8h*)&ldsB[buf][row][colS] = wr[i];
    }
  };
  loadTiles(0);
  stageTiles(0);
  __syncthreads();
  for (int kt = 0; kt < NT; ++kt) {
    if (kt + 1 < NT) loadTiles(kt + 1);
    if (kt + 2 < NT) {
      const int k0 = (kt + 2) * TK;
      if (tid < 128) __builtin_prefetch(xh + (size_t)(mBlock + tid)       * K + k0, 0, 3);
      else           __builtin_prefetch(wh + (size_t)(nBlock + tid - 128) * K + k0, 0, 3);
    }
    wave_mma_step(ldsA[kt & 1], ldsB[kt & 1], acc, waveM, waveN, half, l15);
    __syncthreads();
    if (kt + 1 < NT) {
      stageTiles((kt + 1) & 1);
      __syncthreads();
    }
  }
#endif

  store_epilogue(out, scale, bias, acc, mBlock, nBlock, waveM, waveN, half, l15, N);
}

// ---------------- fallback: fused conversion GEMM (used if workspace too small) ---------
__global__ __launch_bounds__(256)
void qlinear_wmma_f16_fused(const float* __restrict__ x, const int* __restrict__ w,
                            const float* __restrict__ scale, const float* __restrict__ bias,
                            float* __restrict__ out, int M, int N, int K) {
  __shared__ _Float16 ldsA[2][TM][LDST];
  __shared__ _Float16 ldsB[2][TN][LDST];

  const int tid   = threadIdx.x;
  const int lane  = tid & 31;
  const int wave  = tid >> 5;
  const int waveM = (wave & 3) * 32;
  const int waveN = (wave >> 2) * 64;
  const int half  = lane >> 4;
  const int l15   = lane & 15;
  const int mBlock = blockIdx.y * TM;
  const int nBlock = blockIdx.x * TN;

  v8f acc[2][4] = {};
  float4 xa[8];
  int4   wa[8];

  auto loadTiles = [&](int kt) {
    const int k0 = kt * TK;
#pragma unroll
    for (int i = 0; i < 8; ++i) {
      const int c   = tid + i * 256;
      const int row = c >> 4;
      const int col = (c & 15) << 2;
      xa[i] = *(const float4*)(x + (size_t)(mBlock + row) * K + k0 + col);
      wa[i] = *(const int4*  )(w + (size_t)(nBlock + row) * K + k0 + col);
    }
  };
  auto stageTiles = [&](int buf) {
#pragma unroll
    for (int i = 0; i < 8; ++i) {
      const int c   = tid + i * 256;
      const int row = c >> 4;
      const int col = (c & 15) << 2;
      v4h ha = { (_Float16)xa[i].x, (_Float16)xa[i].y,
                 (_Float16)xa[i].z, (_Float16)xa[i].w };
      v4h hb = { (_Float16)(float)wa[i].x, (_Float16)(float)wa[i].y,
                 (_Float16)(float)wa[i].z, (_Float16)(float)wa[i].w };
      *(v4h*)&ldsA[buf][row][col] = ha;
      *(v4h*)&ldsB[buf][row][col] = hb;
    }
  };

  const int NT = K / TK;
  loadTiles(0);
  stageTiles(0);
  __syncthreads();
  for (int kt = 0; kt < NT; ++kt) {
    if (kt + 1 < NT) loadTiles(kt + 1);
    wave_mma_step(ldsA[kt & 1], ldsB[kt & 1], acc, waveM, waveN, half, l15);
    __syncthreads();
    if (kt + 1 < NT) {
      stageTiles((kt + 1) & 1);
      __syncthreads();
    }
  }
  store_epilogue(out, scale, bias, acc, mBlock, nBlock, waveM, waveN, half, l15, N);
}

extern "C" void kernel_launch(void* const* d_in, const int* in_sizes, int n_in,
                              void* d_out, int out_size, void* d_ws, size_t ws_size,
                              hipStream_t stream) {
  const float* x     = (const float*)d_in[0];   // [M, K] fp32
  const int*   w     = (const int*)d_in[1];     // [N, K] int8-in-int32
  const float* scale = (const float*)d_in[2];   // [1]
  const float* bias  = (const float*)d_in[3];   // [N]
  float* out = (float*)d_out;                   // [M, N] fp32

  const int N = in_sizes[3];            // D_OUT
  const int K = in_sizes[1] / N;        // D_IN
  const int M = in_sizes[0] / K;        // B*S

  dim3 grid((unsigned)(N / TN), (unsigned)(M / TM));

  const size_t needW = (size_t)N * K * sizeof(_Float16);
  const size_t needX = (size_t)M * K * sizeof(_Float16);

  if (ws_size >= needW + needX) {
    _Float16* wh = (_Float16*)d_ws;
    _Float16* xh = (_Float16*)((char*)d_ws + needW);
    const long w4 = (long)N * K / 4;
    const long x4 = (long)M * K / 4;
    convert_w_f16<<<(unsigned)((w4 + 255) / 256), 256, 0, stream>>>(w, wh, w4);
    convert_x_f16<<<(unsigned)((x4 + 255) / 256), 256, 0, stream>>>(x, xh, x4);
    qlinear_wmma_f16_pre<<<grid, 256, 0, stream>>>(xh, wh, scale, bias, out, M, N, K);
  } else {
    qlinear_wmma_f16_fused<<<grid, 256, 0, stream>>>(x, w, scale, bias, out, M, N, K);
  }
}
